// Tree_2963527434844
// MI455X (gfx1250) — compile-verified, matrix operand order checked by reading, not run on hardware
//
#include <hip/hip_runtime.h>

typedef __attribute__((ext_vector_type(16))) __bf16 v16bf;
typedef __attribute__((ext_vector_type(8)))  __bf16 v8bf;
typedef __attribute__((ext_vector_type(8)))  float  v8f;
typedef unsigned int u32;
typedef u32 u32x4 __attribute__((ext_vector_type(4)));
typedef u32 u32x8 __attribute__((ext_vector_type(8)));

#define DEPTH   10
#define N_IN    2048
#define N_USED  1024
#define N_LEAF  1024
#define BATCH   16384

#define BLK_M 128
#define BLK_N 128
#define BLK_K 32
#define LDSS  40   // bf16 per LDS tile row: 32 data + 8 pad (80B = 20 banks -> conflict-free b128 frag reads)

// ---------------- Kernel 1: one-hot mask -> gather indices ----------------
__global__ __launch_bounds__(256) void tree_idx_kernel(
    const float* __restrict__ mask, int* __restrict__ idx)
{
  int j = blockIdx.x * 256 + threadIdx.x;
  if (j >= N_USED) return;
  int found = 0;
  for (int i = 0; i < N_IN; ++i) {
    if (mask[(size_t)i * N_USED + j] > 0.5f) found = i;
  }
  idx[j] = found;
}

// ------------- Kernel 1b: split W (f32) into bf16 hi/lo planes ------------
__global__ __launch_bounds__(256) void tree_wsplit_kernel(
    const float* __restrict__ W, __bf16* __restrict__ Whi, __bf16* __restrict__ Wlo)
{
  size_t i = (size_t)blockIdx.x * 1024 + (size_t)threadIdx.x * 4;
  const float4 v = *(const float4*)(W + i);
  const float vv[4] = {v.x, v.y, v.z, v.w};
#pragma unroll
  for (int j = 0; j < 4; ++j) {
    const __bf16 h = (__bf16)vv[j];
    Whi[i + j] = h;
    Wlo[i + j] = (__bf16)(vv[j] - (float)h);
  }
}

// -------------------------- fragment load helper --------------------------
__device__ inline v16bf frag_ld(const __bf16* base, int row, int half) {
  // 16-bit A/B fragment: lanes 0-15 hold K{0..7,16..23}, lanes 16-31 K{8..15,24..31}
  const __bf16* p = base + row * LDSS + half * 8;
  union { v16bf v; v8bf h[2]; } u;
  u.h[0] = *(const v8bf*)(p);
  u.h[1] = *(const v8bf*)(p + 16);
  return u.v;
}

// ----- TDM: 2D bf16 tile (32 x 128 rows, row stride 1024) -> padded LDS ---
// D# per CDNA5 ISA. pad: 16 DWORDs data then 4 DWORDs
// pad per row -> 80B LDS row stride == LDSS.
__device__ inline void tdm_load_tile_bf16(const __bf16* gsrc, u32 lds_byte_off) {
  const unsigned long long ga = (unsigned long long)(uintptr_t)gsrc;
  u32x4 g0;
  g0[0] = 1u;                                        // count=1, user D#
  g0[1] = lds_byte_off;                              // LDS byte address
  g0[2] = (u32)ga;                                   // global_addr[31:0]
  g0[3] = ((u32)(ga >> 32) & 0x01FFFFFFu) | (2u << 30); // addr[56:32] | type=2
  u32x8 g1;
  g1[0] = (1u << 16)        // data_size = 2 bytes
        | (1u << 20)        // pad_enable
        | (3u << 22)        // pad_interval: 16 DWORDs (64B of data)
        | (3u << 25);       // pad_amount: 4 DWORDs (16B pad)
  g1[1] = (u32)(N_USED & 0xFFFF) << 16;              // tensor_dim0[15:0]
  g1[2] = (u32)(N_LEAF & 0xFFFF) << 16;              // dim0 hi=0 | tensor_dim1[15:0]
  g1[3] = (u32)BLK_K << 16;                          // dim1 hi=0 | tile_dim0=32
  g1[4] = (u32)BLK_N;                                // tile_dim1=128, tile_dim2=0
  g1[5] = (u32)N_USED;                               // tensor_dim0_stride lo32
  g1[6] = 0u;                                        // stride hi | dim1_stride lo
  g1[7] = 0u;
  asm volatile("tensor_load_to_lds %0, %1" :: "s"(g0), "s"(g1) : "memory");
}

// ------- Kernel 2 (TDM path): gather-A + TDM-B + bf16x3 WMMA + sigmoid ----
__global__ __launch_bounds__(256) void tree_gemm_tdm_kernel(
    const float* __restrict__ x, const __bf16* __restrict__ Whi,
    const __bf16* __restrict__ Wlo, const float* __restrict__ bias,
    const int* __restrict__ idx, float* __restrict__ dout)
{
  __shared__ __bf16 sAh[BLK_M * LDSS];
  __shared__ __bf16 sAl[BLK_M * LDSS];
  __shared__ __bf16 sBh[BLK_N * LDSS];
  __shared__ __bf16 sBl[BLK_N * LDSS];

  const int tid   = threadIdx.x;
  const int m0    = blockIdx.y * BLK_M;
  const int n0    = blockIdx.x * BLK_N;
  const int lane  = tid & 31;
  const int wave  = tid >> 5;     // 8 waves
  const int wm    = wave >> 2;    // 0..1 : M direction (64 rows each)
  const int wn    = wave & 3;     // 0..3 : N direction (32 cols each)
  const int laneL = lane & 15;
  const int half  = lane >> 4;

  const u32 offBh = (u32)(uintptr_t)(void*)sBh;   // flat-LDS low 32 bits = LDS offset
  const u32 offBl = (u32)(uintptr_t)(void*)sBl;

  v8f acc[4][2] = {};

  const int ka = tid & 31;   // k within chunk
  const int rg = tid >> 5;   // row group for A staging

  for (int k0 = 0; k0 < N_USED; k0 += BLK_K) {
    // ---- kick off async TDM loads of both B planes (overlap with A gather)
    if (wave == 0) {
      tdm_load_tile_bf16(Whi + (size_t)n0 * N_USED + k0, offBh);
      tdm_load_tile_bf16(Wlo + (size_t)n0 * N_USED + k0, offBl);
    }

    // ---- stage A (gathered x columns), split hi/lo bf16 ----
    const int col = idx[k0 + ka];
    const float* xcol = x + col;
#pragma unroll
    for (int i = 0; i < 16; ++i) {
      const int r = rg + 8 * i;                    // 0..127, unique
      const float v = xcol[(size_t)(m0 + r) * N_IN];
      const __bf16 h = (__bf16)v;
      sAh[r * LDSS + ka] = h;
      sAl[r * LDSS + ka] = (__bf16)(v - (float)h);
    }

    if (wave == 0) __builtin_amdgcn_s_wait_tensorcnt(0);
    __syncthreads();

    // ---- fragments ----
    v16bf ah[4], al[4], bh[2], bl[2];
#pragma unroll
    for (int mi = 0; mi < 4; ++mi) {
      const int r = wm * 64 + mi * 16 + laneL;
      ah[mi] = frag_ld(sAh, r, half);
      al[mi] = frag_ld(sAl, r, half);
    }
#pragma unroll
    for (int ni = 0; ni < 2; ++ni) {
      const int r = wn * 32 + ni * 16 + laneL;
      bh[ni] = frag_ld(sBh, r, half);
      bl[ni] = frag_ld(sBl, r, half);
    }

    // ---- 3-term split product: hi*hi + hi*lo + lo*hi ----
#pragma unroll
    for (int mi = 0; mi < 4; ++mi) {
#pragma unroll
      for (int ni = 0; ni < 2; ++ni) {
        acc[mi][ni] = __builtin_amdgcn_wmma_f32_16x16x32_bf16(
            false, ah[mi], false, bh[ni], (short)0, acc[mi][ni], false, false);
        acc[mi][ni] = __builtin_amdgcn_wmma_f32_16x16x32_bf16(
            false, ah[mi], false, bl[ni], (short)0, acc[mi][ni], false, false);
        acc[mi][ni] = __builtin_amdgcn_wmma_f32_16x16x32_bf16(
            false, al[mi], false, bh[ni], (short)0, acc[mi][ni], false, false);
      }
    }
    __syncthreads();
  }

  // ---- epilogue: bias + sigmoid, C/D layout per ISA (VGPR r -> M=r(+8)) ----
  const int nbase = n0 + wn * 32;
#pragma unroll
  for (int ni = 0; ni < 2; ++ni) {
    const int n = nbase + ni * 16 + laneL;
    const float bv = bias[n];
#pragma unroll
    for (int mi = 0; mi < 4; ++mi) {
      const int mb = m0 + wm * 64 + mi * 16 + half * 8;
      const v8f c = acc[mi][ni];
#pragma unroll
      for (int r = 0; r < 8; ++r) {
        float val = c[r] + bv;
        val = 1.0f / (1.0f + __expf(-val));
        dout[(size_t)(mb + r) * N_LEAF + n] = val;
      }
    }
  }
}

// ------ Kernel 2 (fallback, small-ws): fused convert + WMMA + sigmoid -----
__global__ __launch_bounds__(256) void tree_gemm_fb_kernel(
    const float* __restrict__ x, const float* __restrict__ W,
    const float* __restrict__ bias, const int* __restrict__ idx,
    float* __restrict__ dout)
{
  __shared__ __bf16 sAh[BLK_M * LDSS];
  __shared__ __bf16 sAl[BLK_M * LDSS];
  __shared__ __bf16 sBh[BLK_N * LDSS];
  __shared__ __bf16 sBl[BLK_N * LDSS];

  const int tid   = threadIdx.x;
  const int m0    = blockIdx.y * BLK_M;
  const int n0    = blockIdx.x * BLK_N;
  const int lane  = tid & 31;
  const int wave  = tid >> 5;
  const int wm    = wave >> 2;
  const int wn    = wave & 3;
  const int laneL = lane & 15;
  const int half  = lane >> 4;

  v8f acc[4][2] = {};

  const int ka = tid & 31;
  const int rg = tid >> 5;
  const int kb = (tid & 7) * 4;
  const int nb = tid >> 3;

  for (int k0 = 0; k0 < N_USED; k0 += BLK_K) {
    const int col = idx[k0 + ka];
    const float* xcol = x + col;
#pragma unroll
    for (int i = 0; i < 16; ++i) {
      const int r = rg + 8 * i;
      const float v = xcol[(size_t)(m0 + r) * N_IN];
      const __bf16 h = (__bf16)v;
      sAh[r * LDSS + ka] = h;
      sAl[r * LDSS + ka] = (__bf16)(v - (float)h);
    }
#pragma unroll
    for (int i = 0; i < 4; ++i) {
      const int n = nb + 32 * i;
      const float4 wv = *(const float4*)(W + (size_t)(n0 + n) * N_USED + k0 + kb);
      const float vv[4] = {wv.x, wv.y, wv.z, wv.w};
#pragma unroll
      for (int j = 0; j < 4; ++j) {
        const __bf16 h = (__bf16)vv[j];
        sBh[n * LDSS + kb + j] = h;
        sBl[n * LDSS + kb + j] = (__bf16)(vv[j] - (float)h);
      }
    }
    __syncthreads();

    v16bf ah[4], al[4], bh[2], bl[2];
#pragma unroll
    for (int mi = 0; mi < 4; ++mi) {
      const int r = wm * 64 + mi * 16 + laneL;
      ah[mi] = frag_ld(sAh, r, half);
      al[mi] = frag_ld(sAl, r, half);
    }
#pragma unroll
    for (int ni = 0; ni < 2; ++ni) {
      const int r = wn * 32 + ni * 16 + laneL;
      bh[ni] = frag_ld(sBh, r, half);
      bl[ni] = frag_ld(sBl, r, half);
    }
#pragma unroll
    for (int mi = 0; mi < 4; ++mi) {
#pragma unroll
      for (int ni = 0; ni < 2; ++ni) {
        acc[mi][ni] = __builtin_amdgcn_wmma_f32_16x16x32_bf16(
            false, ah[mi], false, bh[ni], (short)0, acc[mi][ni], false, false);
        acc[mi][ni] = __builtin_amdgcn_wmma_f32_16x16x32_bf16(
            false, ah[mi], false, bl[ni], (short)0, acc[mi][ni], false, false);
        acc[mi][ni] = __builtin_amdgcn_wmma_f32_16x16x32_bf16(
            false, al[mi], false, bh[ni], (short)0, acc[mi][ni], false, false);
      }
    }
    __syncthreads();
  }

  const int nbase = n0 + wn * 32;
#pragma unroll
  for (int ni = 0; ni < 2; ++ni) {
    const int n = nbase + ni * 16 + laneL;
    const float bv = bias[n];
#pragma unroll
    for (int mi = 0; mi < 4; ++mi) {
      const int mb = m0 + wm * 64 + mi * 16 + half * 8;
      const v8f c = acc[mi][ni];
#pragma unroll
      for (int r = 0; r < 8; ++r) {
        float val = c[r] + bv;
        val = 1.0f / (1.0f + __expf(-val));
        dout[(size_t)(mb + r) * N_LEAF + n] = val;
      }
    }
  }
}

// ---------------- Kernel 3: soft tree routing (in-place on d) -------------
__global__ __launch_bounds__(256) void tree_route_kernel(float* __restrict__ d)
{
  __shared__ float s[N_LEAF];
  const int row = blockIdx.x;
  float* dr = d + (size_t)row * N_LEAF;
  for (int i = threadIdx.x; i < N_LEAF; i += 256) s[i] = dr[i];
  __syncthreads();

  float out[4];
#pragma unroll
  for (int q = 0; q < 4; ++q) {
    const int l = threadIdx.x + 256 * q;
    float mu = 1.0f;
#pragma unroll
    for (int n = 0; n < DEPTH; ++n) {
      const int node = (1 << n) + (l >> (DEPTH - n));
      const float dv = s[node];
      mu *= ((l >> (DEPTH - 1 - n)) & 1) ? (1.0f - dv) : dv;
    }
    out[q] = mu;
  }
  __syncthreads();
#pragma unroll
  for (int q = 0; q < 4; ++q) dr[threadIdx.x + 256 * q] = out[q];
}

// --------------------------------- launch ---------------------------------
extern "C" void kernel_launch(void* const* d_in, const int* in_sizes, int n_in,
                              void* d_out, int out_size, void* d_ws, size_t ws_size,
                              hipStream_t stream) {
  const float* x    = (const float*)d_in[0];   // [16384, 2048]
  const float* mask = (const float*)d_in[1];   // [2048, 1024] one-hot cols
  const float* W    = (const float*)d_in[2];   // [1024, 1024]
  const float* b    = (const float*)d_in[3];   // [1024]
  float* out = (float*)d_out;                  // [16384, 1024]

  char* ws = (char*)d_ws;
  int* idx = (int*)ws;                                          // 4 KB
  const size_t wBytes = (size_t)N_LEAF * N_USED * sizeof(__bf16); // 2 MB
  __bf16* Whi = (__bf16*)(ws + 4096);
  __bf16* Wlo = (__bf16*)(ws + 4096 + wBytes);
  const size_t ws_needed = 4096 + 2 * wBytes;

  tree_idx_kernel<<<dim3((N_USED + 255) / 256), dim3(256), 0, stream>>>(mask, idx);

  if (ws_size >= ws_needed) {
    tree_wsplit_kernel<<<dim3((N_LEAF * N_USED) / 1024), dim3(256), 0, stream>>>(
        W, Whi, Wlo);
    tree_gemm_tdm_kernel<<<dim3(N_LEAF / BLK_N, BATCH / BLK_M), dim3(256), 0, stream>>>(
        x, Whi, Wlo, b, idx, out);
  } else {
    tree_gemm_fb_kernel<<<dim3(N_LEAF / BLK_N, BATCH / BLK_M), dim3(256), 0, stream>>>(
        x, W, b, idx, out);
  }
  tree_route_kernel<<<dim3(BATCH), dim3(256), 0, stream>>>(out);
}
